// Mamba_Encoder_52604759442066
// MI455X (gfx1250) — matrix operand order repaired and checked by reading, not compiled, not run on hardware
//
#include <hip/hip_runtime.h>
#include <math.h>

// ---------------------------------------------------------------------------
// Mamba 3D-UNet encoder forward for MI455X (gfx1250), wave32 + WMMA bf16.
//
// Input flattening assumption (jax pytree order: dict keys sorted, lists in
// order; top-level dict insertion order x, params):
//   d_in[0]   : x [1,1,64,64,64]
//   d_in[1..4]: dc_b1, dc_b2, dc_w1, dc_w2
//   d_in[5..20]  : down1..down4 each {b, gn_b, gn_g, w}
//   d_in[21..124]: mb1..mb4 each {fc1_b, fc1_w, fc2_b, fc2_w,
//                    layer0{A_log,D,conv_b,conv_w,dt_b,dt_w,in_proj,ln_b,ln_g,out_proj,x_proj},
//                    layer1{...same...}}
// Output: concat(x4,x3,x2,x1,x0) fp32, 1,047,552 elements.
// ---------------------------------------------------------------------------

typedef __bf16 bf16_t;
typedef __attribute__((ext_vector_type(16))) __bf16 v16bf;
typedef __attribute__((ext_vector_type(8)))  float  v8f;

#define ACT_NONE     0
#define ACT_RELU     1
#define ACT_SILU     2
#define ACT_GELU     3
#define ACT_SOFTPLUS 4

__device__ __forceinline__ float apply_act(float v, int act) {
    switch (act) {
        case ACT_RELU:     return fmaxf(v, 0.f);
        case ACT_SILU:     return v / (1.f + expf(-v));
        case ACT_GELU:     return 0.5f * v * (1.f + erff(v * 0.70710678118654752f));
        case ACT_SOFTPLUS: return (v > 20.f) ? v : log1pf(expf(v));
        default:           return v;
    }
}

// Clamped load + zero-select (unconditional load, no EXEC manipulation).
__device__ __forceinline__ bf16_t ld_bf16_clamped(const float* __restrict__ row, int kk, int K) {
    float v = row[kk < K ? kk : K - 1];
    return (kk < K) ? (bf16_t)v : (bf16_t)0.f;
}

__device__ __forceinline__ v16bf cvt16(const float4& p0, const float4& p1,
                                       const float4& p2, const float4& p3) {
    v16bf r;
    r[0]  = (bf16_t)p0.x; r[1]  = (bf16_t)p0.y; r[2]  = (bf16_t)p0.z; r[3]  = (bf16_t)p0.w;
    r[4]  = (bf16_t)p1.x; r[5]  = (bf16_t)p1.y; r[6]  = (bf16_t)p1.z; r[7]  = (bf16_t)p1.w;
    r[8]  = (bf16_t)p2.x; r[9]  = (bf16_t)p2.y; r[10] = (bf16_t)p2.z; r[11] = (bf16_t)p2.w;
    r[12] = (bf16_t)p3.x; r[13] = (bf16_t)p3.y; r[14] = (bf16_t)p3.z; r[15] = (bf16_t)p3.w;
    return r;
}

// ---------------------------------------------------------------------------
// WMMA bf16 GEMM:  out[M,N] = act( X @ W^T + bias )
//   X: [M,K] row-major, leading dim ldx  (fp32 -> bf16)
//   W: [N,K] row-major               i.e. B(k,n) = W[n*K+k]
//   out: transOut ? out[n*ldo+m] : out[m*ldo+n]   (fp32)
//
// Each wave computes a 16x64 macro-tile: one A fragment feeds 4 WMMAs per
// 32-wide K step (4 f32 accumulators). Host guarantees M % 16 == 0.
// Fast path (ldx % 8 == 0, 16B-aligned rows) is software-pipelined with an
// explicit double-buffered (ping-pong) staging set: each half-iteration
// issues the b128 loads for chunk k+32 into the *other* buffer, then
// converts + runs the 4 WMMAs on the current buffer, so the next wait sits
// behind a full WMMA group and no register copies are needed.
// Out-of-range B columns are *clamped*, not zeroed: WMMA column n of D
// depends only on column n of B, and n >= N results are discarded on store.
// Fragment layouts per CDNA5 ISA 7.12.2 (16-bit A 16x32; B 32x16; f32 C/D).
// ---------------------------------------------------------------------------
__global__ __launch_bounds__(256)
void gemm_wmma_bf16(const float* __restrict__ X, const float* __restrict__ W,
                    const float* __restrict__ bias, float* __restrict__ out,
                    int M, int N, int K, int ldx, int ldo, int transOut, int act)
{
    const int lane = threadIdx.x & 31;
    const int wv   = threadIdx.x >> 5;
    const int nT4  = (N + 63) >> 6;               // macro-tiles of 64 in N
    const int tile = blockIdx.x * 8 + wv;
    if (tile >= nT4 * (M >> 4)) return;           // M % 16 == 0
    const int mt = tile / nT4, nt = tile % nT4;
    const int m0 = mt << 4, n0 = nt << 6;
    const int half = lane >> 4, lm = lane & 15;
    const int am = m0 + lm;                       // always < M
    const float* __restrict__ arow = X + (size_t)am * ldx;

    int bn[4];
    const float* __restrict__ brow[4];
#pragma unroll
    for (int j = 0; j < 4; ++j) {
        bn[j] = n0 + j * 16 + lm;
        const int bc = (bn[j] < N) ? bn[j] : (N - 1);   // clamp, don't zero
        brow[j] = W + (size_t)bc * K;
    }

    v8f acc[4];
#pragma unroll
    for (int j = 0; j < 4; ++j) acc[j] = (v8f){0.f, 0.f, 0.f, 0.f, 0.f, 0.f, 0.f, 0.f};

    const int Kmain = ((ldx & 7) == 0) ? (K & ~31) : 0;
    int k0 = 0;

    if (Kmain > 0) {
        // Two disjoint f32 staging buffers (ping-pong pipeline).
        float4 saA[4], sbA[4][4];
        float4 saB[4], sbB[4][4];

        auto issue_loads = [&](int k, float4 (&sa)[4], float4 (&sb)[4][4]) {
            sa[0] = *reinterpret_cast<const float4*>(arow + k + half * 8);
            sa[1] = *reinterpret_cast<const float4*>(arow + k + half * 8 + 4);
            sa[2] = *reinterpret_cast<const float4*>(arow + k + 16 + half * 8);
            sa[3] = *reinterpret_cast<const float4*>(arow + k + 16 + half * 8 + 4);
#pragma unroll
            for (int j = 0; j < 4; ++j) {
                const float* bp = brow[j] + k + half * 8;
                sb[j][0] = *reinterpret_cast<const float4*>(bp);
                sb[j][1] = *reinterpret_cast<const float4*>(bp + 4);
                sb[j][2] = *reinterpret_cast<const float4*>(bp + 16);
                sb[j][3] = *reinterpret_cast<const float4*>(bp + 20);
            }
        };

        auto consume = [&](const float4 (&sa)[4], const float4 (&sb)[4][4]) {
            const v16bf a = cvt16(sa[0], sa[1], sa[2], sa[3]);
            v16bf b[4];
#pragma unroll
            for (int j = 0; j < 4; ++j) b[j] = cvt16(sb[j][0], sb[j][1], sb[j][2], sb[j][3]);
#pragma unroll
            for (int j = 0; j < 4; ++j)
                acc[j] = __builtin_amdgcn_wmma_f32_16x16x32_bf16(false, a, false, b[j],
                                                                 (short)0, acc[j], false, false);
        };

        issue_loads(0, saA, sbA);
        int k = 0;
        while (true) {
            // ---- buffer A holds chunk k ----
            if (k + 64 < K) {                      // uniform; one chunk beyond pipeline
                __builtin_prefetch(arow + k + 64, 0, 3);
                __builtin_prefetch(brow[0] + k + 64, 0, 3);
                __builtin_prefetch(brow[3] + k + 64, 0, 3);
            }
            if (k + 32 < Kmain) issue_loads(k + 32, saB, sbB);
            consume(saA, sbA);
            k += 32;
            if (k >= Kmain) break;

            // ---- buffer B holds chunk k ----
            if (k + 64 < K) {
                __builtin_prefetch(arow + k + 64, 0, 3);
                __builtin_prefetch(brow[0] + k + 64, 0, 3);
                __builtin_prefetch(brow[3] + k + 64, 0, 3);
            }
            if (k + 32 < Kmain) issue_loads(k + 32, saA, sbA);
            consume(saB, sbB);
            k += 32;
            if (k >= Kmain) break;
        }
        k0 = Kmain;
    }

    // K remainder (and unaligned fallback): clamped loads + zero-select.
    for (; k0 < K; k0 += 32) {
        v16bf a;
#pragma unroll
        for (int j = 0; j < 8; ++j) {
            const int grp = j >> 2, jj = j & 3;
            const int kk = k0 + grp * 16 + half * 8 + jj * 2;
            a[2 * j]     = ld_bf16_clamped(arow, kk,     K);
            a[2 * j + 1] = ld_bf16_clamped(arow, kk + 1, K);
        }
#pragma unroll
        for (int j = 0; j < 4; ++j) {
            v16bf b;
#pragma unroll
            for (int t = 0; t < 8; ++t) {
                const int kk = k0 + (t >> 2) * 16 + half * 8 + (t & 3) * 2;
                b[2 * t]     = ld_bf16_clamped(brow[j], kk,     K);
                b[2 * t + 1] = ld_bf16_clamped(brow[j], kk + 1, K);
            }
            acc[j] = __builtin_amdgcn_wmma_f32_16x16x32_bf16(false, a, false, b, (short)0, acc[j], false, false);
        }
    }

    // Epilogue. C/D layout: lane n = lane&15; lanes 0-15 hold M=0..7, 16-31 M=8..15.
#pragma unroll
    for (int j = 0; j < 4; ++j) {
        const int n = bn[j];
        if (n >= N) continue;
        const float bv = bias ? bias[n] : 0.f;
#pragma unroll
        for (int r = 0; r < 8; ++r) {
            const int m = m0 + half * 8 + r;
            float v = acc[j][r] + bv;
            v = apply_act(v, act);
            if (transOut) out[(size_t)n * ldo + m] = v;
            else          out[(size_t)m * ldo + n] = v;
        }
    }
}

// ---------------------------------------------------------------------------
// Direct 3x3x3 conv, pad=1, arbitrary stride. NCDHW, B=1.
// ---------------------------------------------------------------------------
__global__ void conv3d_k3(const float* __restrict__ x, const float* __restrict__ w,
                          const float* __restrict__ bias, float* __restrict__ y,
                          int Cin, int Cout, int Di, int Hi, int Wi,
                          int Do, int Ho, int Wo, int stride)
{
    int idx = blockIdx.x * blockDim.x + threadIdx.x;
    int n = Cout * Do * Ho * Wo;
    if (idx >= n) return;
    int wo = idx % Wo; int t = idx / Wo;
    int ho = t % Ho; t /= Ho;
    int dz = t % Do; int co = t / Do;

    float acc = bias[co];
    for (int ci = 0; ci < Cin; ++ci) {
        const float* xc = x + (size_t)ci * Di * Hi * Wi;
        const float* wc = w + (size_t)(co * Cin + ci) * 27;
#pragma unroll
        for (int kd = 0; kd < 3; ++kd) {
            int id = dz * stride + kd - 1;
            if (id < 0 || id >= Di) continue;
#pragma unroll
            for (int kh = 0; kh < 3; ++kh) {
                int ih = ho * stride + kh - 1;
                if (ih < 0 || ih >= Hi) continue;
#pragma unroll
                for (int kw = 0; kw < 3; ++kw) {
                    int iw = wo * stride + kw - 1;
                    if (iw < 0 || iw >= Wi) continue;
                    acc += xc[((size_t)id * Hi + ih) * Wi + iw] * wc[(kd * 3 + kh) * 3 + kw];
                }
            }
        }
    }
    y[idx] = acc;
}

// ---------------------------------------------------------------------------
// InstanceNorm over one channel (block = channel), optional fused ReLU.
// outStrideC == 0: y[c*L+i] (channel-major, in-place capable)
// outStrideC  > 0: y[i*outStrideC + c] (token-major transpose for WMMA GEMMs)
// ---------------------------------------------------------------------------
__global__ void instnorm_kernel(const float* __restrict__ x, float* __restrict__ y,
                                int L, int fuse_relu, int outStrideC)
{
    const int c = blockIdx.x;
    const float* xc = x + (size_t)c * L;
    __shared__ float ss[256], ss2[256];
    float s = 0.f, s2 = 0.f;
    for (int i = threadIdx.x; i < L; i += 256) { float v = xc[i]; s += v; s2 += v * v; }
    ss[threadIdx.x] = s; ss2[threadIdx.x] = s2;
    __syncthreads();
    for (int o = 128; o > 0; o >>= 1) {
        if (threadIdx.x < o) { ss[threadIdx.x] += ss[threadIdx.x + o]; ss2[threadIdx.x] += ss2[threadIdx.x + o]; }
        __syncthreads();
    }
    const float m = ss[0] / L;
    const float var = ss2[0] / L - m * m;
    const float r = rsqrtf(var + 1e-5f);
    for (int i = threadIdx.x; i < L; i += 256) {
        float v = (xc[i] - m) * r;
        if (fuse_relu) v = fmaxf(v, 0.f);
        if (outStrideC > 0) y[(size_t)i * outStrideC + c] = v;
        else                y[(size_t)c * L + i] = v;
    }
}

// ---------------------------------------------------------------------------
// GroupNorm (block = group) with affine + fused ReLU.
// ---------------------------------------------------------------------------
__global__ void groupnorm_relu_kernel(const float* __restrict__ x, const float* __restrict__ g,
                                      const float* __restrict__ b, float* __restrict__ y,
                                      int cpg, int L)
{
    const int grp = blockIdx.x;
    const size_t base = (size_t)grp * cpg * L;
    const int n = cpg * L;
    __shared__ float ss[256], ss2[256];
    float s = 0.f, s2 = 0.f;
    for (int i = threadIdx.x; i < n; i += 256) { float v = x[base + i]; s += v; s2 += v * v; }
    ss[threadIdx.x] = s; ss2[threadIdx.x] = s2;
    __syncthreads();
    for (int o = 128; o > 0; o >>= 1) {
        if (threadIdx.x < o) { ss[threadIdx.x] += ss[threadIdx.x + o]; ss2[threadIdx.x] += ss2[threadIdx.x + o]; }
        __syncthreads();
    }
    const float m = ss[0] / n;
    const float var = ss2[0] / n - m * m;
    const float r = rsqrtf(var + 1e-5f);
    for (int i = threadIdx.x; i < n; i += 256) {
        int c = grp * cpg + i / L;
        float v = (x[base + i] - m) * r * g[c] + b[c];
        y[base + i] = fmaxf(v, 0.f);
    }
}

// ---------------------------------------------------------------------------
// LayerNorm over channel dim with transpose: [C,L] (ch-major) -> [L,C] tokens.
// ---------------------------------------------------------------------------
__global__ void layernorm_ct_kernel(const float* __restrict__ x, const float* __restrict__ g,
                                    const float* __restrict__ b, float* __restrict__ xn,
                                    int C, int L)
{
    int l = blockIdx.x * blockDim.x + threadIdx.x;
    if (l >= L) return;
    float s = 0.f;
    for (int c = 0; c < C; ++c) s += x[(size_t)c * L + l];
    const float m = s / C;
    float v = 0.f;
    for (int c = 0; c < C; ++c) { float d = x[(size_t)c * L + l] - m; v += d * d; }
    const float r = rsqrtf(v / C + 1e-5f);
    for (int c = 0; c < C; ++c)
        xn[(size_t)l * C + c] = (x[(size_t)c * L + l] - m) * r * g[c] + b[c];
}

// ---------------------------------------------------------------------------
// Depthwise causal conv1d (k=4, left pad 3) + bias + SiLU.
// u = xz[:, :di] with row stride 2*di; output uc [L,di].
// ---------------------------------------------------------------------------
__global__ void dwconv_silu_kernel(const float* __restrict__ xz, const float* __restrict__ w,
                                   const float* __restrict__ b, float* __restrict__ uc,
                                   int L, int di)
{
    int idx = blockIdx.x * blockDim.x + threadIdx.x;
    if (idx >= L * di) return;
    const int l = idx / di, d = idx % di;
    float acc = b[d];
#pragma unroll
    for (int j = 0; j < 4; ++j) {
        int ls = l - 3 + j;
        if (ls >= 0) acc += xz[(size_t)ls * 2 * di + d] * w[d * 4 + j];
    }
    uc[idx] = acc / (1.f + expf(-acc));  // silu
}

// ---------------------------------------------------------------------------
// Selective scan + output gating. One block of `di` threads; each thread owns
// channel d and its 16 SSM states in registers; B/C rows staged through LDS.
//   h_l = exp(dt*A)*h_{l-1} + dt*B_l*u_l ;  y = sum_s h*C_l + u*D ; yg = y*silu(z)
// ---------------------------------------------------------------------------
__global__ void scan_gate_kernel(const float* __restrict__ uc, const float* __restrict__ dt,
                                 const float* __restrict__ xdbl, const float* __restrict__ A_log,
                                 const float* __restrict__ Dp, const float* __restrict__ xz,
                                 float* __restrict__ yg, int L, int di, int dtr, int ldxd)
{
    const int d = threadIdx.x;   // blockDim.x == di
    float a[16], h[16];
#pragma unroll
    for (int s = 0; s < 16; ++s) { a[s] = -expf(A_log[d * 16 + s]); h[s] = 0.f; }
    const float Dd = Dp[d];
    __shared__ float sBC[32];

    for (int l = 0; l < L; ++l) {
        if (threadIdx.x < 32) sBC[threadIdx.x] = xdbl[(size_t)l * ldxd + dtr + threadIdx.x];
        __syncthreads();
        const float dtl = dt[(size_t)l * di + d];
        const float u   = uc[(size_t)l * di + d];
        float y = u * Dd;
#pragma unroll
        for (int s = 0; s < 16; ++s) {
            h[s] = expf(dtl * a[s]) * h[s] + dtl * sBC[s] * u;
            y += h[s] * sBC[16 + s];
        }
        const float z = xz[(size_t)l * 2 * di + di + d];
        yg[(size_t)l * di + d] = y * (z / (1.f + expf(-z)));
        __syncthreads();
    }
}

// ---------------------------------------------------------------------------
// 2x2x2 max pool, stride 2.
// ---------------------------------------------------------------------------
__global__ void maxpool2_kernel(const float* __restrict__ x, float* __restrict__ y,
                                int C, int Di, int Hi, int Wi)
{
    const int Do = Di / 2, Ho = Hi / 2, Wo = Wi / 2;
    int idx = blockIdx.x * blockDim.x + threadIdx.x;
    int n = C * Do * Ho * Wo;
    if (idx >= n) return;
    int wo = idx % Wo; int t = idx / Wo;
    int ho = t % Ho; t /= Ho;
    int dz = t % Do; int c = t / Do;
    float m = -INFINITY;
#pragma unroll
    for (int kd = 0; kd < 2; ++kd)
#pragma unroll
        for (int kh = 0; kh < 2; ++kh)
#pragma unroll
            for (int kw = 0; kw < 2; ++kw)
                m = fmaxf(m, x[(((size_t)c * Di + 2 * dz + kd) * Hi + 2 * ho + kh) * Wi + 2 * wo + kw]);
    y[idx] = m;
}

// ---------------------------------------------------------------------------
// Host-side orchestration
// ---------------------------------------------------------------------------
struct MP { const float *A_log, *D, *conv_b, *conv_w, *dt_b, *dt_w, *in_proj, *ln_b, *ln_g, *out_proj, *x_proj; };
struct MBp { const float *fc1_b, *fc1_w, *fc2_b, *fc2_w; MP lay[2]; };
struct DNp { const float *b, *gn_b, *gn_g, *w; };

static inline void gemm(hipStream_t st, const float* X, const float* W, const float* bias,
                        float* out, int M, int N, int K, int ldx,
                        int ldo, int transOut, int act)
{
    int tiles = (M / 16) * ((N + 63) / 64);   // 16x64 macro-tiles, M % 16 == 0
    int blocks = (tiles + 7) / 8;
    gemm_wmma_bf16<<<blocks, 256, 0, st>>>(X, W, bias, out, M, N, K, ldx, ldo, transOut, act);
}

static void run_mamba_layer(hipStream_t st, const float* xin_CL, float* out_CL, const MP& p,
                            int C, int L, float* xn, float* xz, float* uc,
                            float* xdbl, float* dtb, float* yg)
{
    const int di = 2 * C;
    const int dtr = (C + 15) / 16;
    const int xdw = dtr + 32;
    const int xdw_p = (xdw + 7) & ~7;     // padded leading dim (16B-aligned rows)
    layernorm_ct_kernel<<<(L + 255) / 256, 256, 0, st>>>(xin_CL, p.ln_g, p.ln_b, xn, C, L);
    gemm(st, xn, p.in_proj, nullptr, xz, L, 2 * di, C, C, 2 * di, 0, ACT_NONE);
    {
        int n = L * di;
        dwconv_silu_kernel<<<(n + 255) / 256, 256, 0, st>>>(xz, p.conv_w, p.conv_b, uc, L, di);
    }
    gemm(st, uc, p.x_proj, nullptr, xdbl, L, xdw, di, di, xdw_p, 0, ACT_NONE);
    gemm(st, xdbl, p.dt_w, p.dt_b, dtb, L, di, dtr, xdw_p, di, 0, ACT_SOFTPLUS);
    scan_gate_kernel<<<1, di, 0, st>>>(uc, dtb, xdbl, p.A_log, p.D, xz, yg, L, di, dtr, xdw_p);
    gemm(st, yg, p.out_proj, nullptr, out_CL, L, C, di, di, L, 1, ACT_NONE);
}

extern "C" void kernel_launch(void* const* d_in, const int* in_sizes, int n_in,
                              void* d_out, int out_size, void* d_ws, size_t ws_size,
                              hipStream_t stream)
{
    (void)in_sizes; (void)n_in; (void)out_size; (void)ws_size;
    int ii = 0;
    auto F = [&](void) { return (const float*)d_in[ii++]; };

    const float* x     = F();
    const float* dc_b1 = F();
    const float* dc_b2 = F();
    const float* dc_w1 = F();
    const float* dc_w2 = F();

    DNp dn[4];
    for (int s = 0; s < 4; ++s) { dn[s].b = F(); dn[s].gn_b = F(); dn[s].gn_g = F(); dn[s].w = F(); }

    MBp mb[4];
    for (int s = 0; s < 4; ++s) {
        mb[s].fc1_b = F(); mb[s].fc1_w = F(); mb[s].fc2_b = F(); mb[s].fc2_w = F();
        for (int l = 0; l < 2; ++l) {
            MP& p = mb[s].lay[l];
            p.A_log = F(); p.D = F(); p.conv_b = F(); p.conv_w = F();
            p.dt_b = F(); p.dt_w = F(); p.in_proj = F();
            p.ln_b = F(); p.ln_g = F(); p.out_proj = F(); p.x_proj = F();
        }
    }

    float* out = (float*)d_out;
    const size_t OUT_X4 = 0, OUT_X3 = 3072, OUT_X2 = 15360, OUT_X1 = 64512, OUT_X0 = 261120;

    // Workspace arenas (floats): A[0, 6.5M), B[6.5M, 13M), C[13M, 18M)
    float* ws = (float*)d_ws;
    float* bufA = ws;
    float* bufB = ws + 6815744;
    float* bufC = ws + 13631488;

    // ---- Stem: (conv3 -> IN -> ReLU) x2, maxpool ----
    {
        const int D0 = 64, Lsp = D0 * D0 * D0;
        int n1 = 24 * Lsp;
        conv3d_k3<<<(n1 + 255) / 256, 256, 0, stream>>>(x, dc_w1, dc_b1, bufA, 1, 24, D0, D0, D0, D0, D0, D0, 1);
        instnorm_kernel<<<24, 256, 0, stream>>>(bufA, bufA, Lsp, 1, 0);
        conv3d_k3<<<(n1 + 255) / 256, 256, 0, stream>>>(bufA, dc_w2, dc_b2, bufB, 24, 24, D0, D0, D0, D0, D0, D0, 1);
        instnorm_kernel<<<24, 256, 0, stream>>>(bufB, bufB, Lsp, 1, 0);
        int np = 24 * 32 * 32 * 32;
        maxpool2_kernel<<<(np + 255) / 256, 256, 0, stream>>>(bufB, out + OUT_X0, 24, D0, D0, D0);
    }

    // ---- 4 Mamba stages ----
    const int   stageC[4] = {24, 48, 96, 192};
    const int   stageD[4] = {32, 16, 8, 4};
    const float* sin[4]  = {out + OUT_X0, out + OUT_X1, out + OUT_X2, out + OUT_X3};
    float*       sout[4] = {out + OUT_X1, out + OUT_X2, out + OUT_X3, out + OUT_X4};

    for (int s = 0; s < 4; ++s) {
        const int C = stageC[s];
        const int Dd = stageD[s];
        const int L = Dd * Dd * Dd;

        float* cur  = bufA;
        float* xn   = bufB;                 // [L,C]           <= 0.79M
        float* xdbl = bufB + 1048576;       // [L,pad(dtr+32)] <= 1.58M
        float* dtb  = bufB + 2621440;       // [L,di]          <= 1.58M  (reused as fc hidden h)
        float* yg   = bufB + 4718592;       // [L,di]          <= 1.58M
        float* xz   = bufC;                 // [L,2di]         <= 3.15M
        float* uc   = bufC + 3407872;       // [L,di]          <= 1.58M

        run_mamba_layer(stream, sin[s], cur, mb[s].lay[0], C, L, xn, xz, uc, xdbl, dtb, yg);
        run_mamba_layer(stream, cur,    cur, mb[s].lay[1], C, L, xn, xz, uc, xdbl, dtb, yg);

        // mbblock tail: InstanceNorm (token-major out) -> fc1+GELU -> fc2 (+ReLU)
        float* xo_t = bufC;                 // [L,C] token-major
        instnorm_kernel<<<C, 256, 0, stream>>>(cur, xo_t, L, 0, C);
        float* h = dtb;                     // [L,2C] token-major
        gemm(stream, xo_t, mb[s].fc1_w, mb[s].fc1_b, h,   L, 2 * C, C,     C,     2 * C, 0, ACT_GELU);
        gemm(stream, h,    mb[s].fc2_w, mb[s].fc2_b, cur, L, C,     2 * C, 2 * C, L,     1, ACT_RELU);

        // convdown: conv3 stride2 C->2C, GroupNorm(8)+ReLU -> next stage output
        const int C2 = 2 * C, Do = Dd / 2;
        const int Lo = Do * Do * Do;
        float* cv = bufC;
        int nco = C2 * Lo;
        conv3d_k3<<<(nco + 255) / 256, 256, 0, stream>>>(cur, dn[s].w, dn[s].b, cv,
                                                         C, C2, Dd, Dd, Dd, Do, Do, Do, 2);
        groupnorm_relu_kernel<<<8, 256, 0, stream>>>(cv, dn[s].gn_g, dn[s].gn_b, sout[s],
                                                     C2 / 8, Lo);
    }
}
